// TransformerEncoderLayer_64244120813942
// MI455X (gfx1250) — matrix operand order repaired and checked by reading, not compiled, not run on hardware
//
#include <hip/hip_runtime.h>
#include <hip/hip_bf16.h>
#include <math.h>

// ---------------------------------------------------------------------------
// Transformer encoder layer, MI455X (gfx1250), wave32 + WMMA bf16.
// B=4 N=2048 D=1024 A=1024 H=16 HD=64 DFF=4096, scale = D^-0.5 = 1/32.
// ---------------------------------------------------------------------------

#define B_   4
#define N_   2048
#define D_   1024
#define A_   1024
#define H_   16
#define HD_  64
#define DFF_ 4096
#define M_   (B_ * N_)          // 8192 rows
#define SCALE_ 0.03125f         // 1024^-0.5
#define EPS_ 1e-5f

typedef __bf16 v16bf __attribute__((ext_vector_type(16)));
typedef float  v8f   __attribute__((ext_vector_type(8)));
typedef unsigned short u16;

// float -> bf16 (round to nearest even), raw bits
__device__ __forceinline__ u16 f2bf(float f) {
    unsigned int u = __float_as_uint(f);
    u = (u + 0x7FFFu + ((u >> 16) & 1u)) >> 16;
    return (u16)u;
}

union FragU { v16bf v; uint4 u[2]; };

// Build a 16-element bf16 fragment from two contiguous 16-byte groups.
__device__ __forceinline__ v16bf ld2(const u16* p0, const u16* p1) {
    FragU f;
    f.u[0] = *(const uint4*)p0;
    f.u[1] = *(const uint4*)p1;
    return f.v;
}

__device__ __forceinline__ v8f wmma_bf16(v16bf a, v16bf b, v8f c) {
    return __builtin_amdgcn_wmma_f32_16x16x32_bf16(
        /*neg_a=*/false, a, /*neg_b=*/false, b,
        /*c_mod=*/(short)0, c, /*reuse_a=*/false, /*reuse_b=*/false);
}

// Async DMA: 16 bytes global -> LDS, tracked by ASYNCcnt (gfx1250).
// VDST = per-lane LDS byte offset, VADDR = 64-bit global address.
__device__ __forceinline__ void async_copy16(unsigned lds_off, const void* gaddr) {
    asm volatile("global_load_async_to_lds_b128 %0, %1, off"
                 :: "v"(lds_off), "v"(gaddr) : "memory");
}

// ---------------------------------------------------------------------------
// fp32 -> bf16 elementwise (weights)
// ---------------------------------------------------------------------------
__global__ void cvt_bf16(const float* __restrict__ in, u16* __restrict__ out, int n) {
    int i = (blockIdx.x * blockDim.x + threadIdx.x) * 4;
    if (i < n) {
        float4 v = *(const float4*)(in + i);
        out[i + 0] = f2bf(v.x);
        out[i + 1] = f2bf(v.y);
        out[i + 2] = f2bf(v.z);
        out[i + 3] = f2bf(v.w);
    }
}

// ---------------------------------------------------------------------------
// LayerNorm: one block per row of D_=1024; 256 threads x float4.
// Writes bf16 normalized activations.
// ---------------------------------------------------------------------------
__global__ __launch_bounds__(256)
void ln_bf16(const float* __restrict__ x, const float* __restrict__ g,
             const float* __restrict__ be, u16* __restrict__ out) {
    __shared__ float rs[256], rq[256];
    const int row = blockIdx.x, tid = threadIdx.x;
    const float4 v = ((const float4*)(x + (size_t)row * D_))[tid];
    rs[tid] = v.x + v.y + v.z + v.w;
    rq[tid] = v.x * v.x + v.y * v.y + v.z * v.z + v.w * v.w;
    __syncthreads();
    for (int off = 128; off > 0; off >>= 1) {
        if (tid < off) { rs[tid] += rs[tid + off]; rq[tid] += rq[tid + off]; }
        __syncthreads();
    }
    const float mean = rs[0] * (1.0f / D_);
    const float var  = rq[0] * (1.0f / D_) - mean * mean;
    const float rinv = rsqrtf(var + EPS_);
    const int c = tid * 4;
    u16* o = out + (size_t)row * D_ + c;
    o[0] = f2bf((v.x - mean) * rinv * g[c + 0] + be[c + 0]);
    o[1] = f2bf((v.y - mean) * rinv * g[c + 1] + be[c + 1]);
    o[2] = f2bf((v.z - mean) * rinv * g[c + 2] + be[c + 2]);
    o[3] = f2bf((v.w - mean) * rinv * g[c + 3] + be[c + 3]);
}

// ---------------------------------------------------------------------------
// WMMA GEMM:  Y[M,Nout] = act( X[M,K](bf16) @ W[Nout,K]^T (bf16) + bias ) (+ res)
// Block = 256 threads = 8 waves, block tile 128x128.
// Wave (wm in 0..1, wn in 0..3) owns a 64x32 tile = 4x2 accumulator tiles.
//
// A tile (128x32 bf16, 8KB) is double-buffered in LDS and filled with
// global_load_async_to_lds_b128; the ASYNCcnt pipeline overlaps the next
// tile's DMA with the current tile's 8 WMMAs. B fragments (weight rows,
// B[c][n] = W[n][c]) read 16 contiguous bf16 straight from global (the whole
// bf16 weight set is 24MB and lives in the 192MB L2).
//
// A-fragment groups at 16*half and 16*half+8 (contiguous k = 16*half + s)
// B-fragment groups at k0+16*half, +8        (same contiguous k order as A)
// ---------------------------------------------------------------------------
template <bool RELU>
__global__ __launch_bounds__(256)
void gemm_wmma(const u16* __restrict__ X, const u16* __restrict__ W,
               const float* __restrict__ bias, const float* __restrict__ res,
               float* __restrict__ outf, u16* __restrict__ outb,
               int M, int Nout, int K) {
    __shared__ u16 Atile[2][128][32];   // 2 x 8KB

    const int tid  = threadIdx.x;
    const int lane = tid & 31;
    const int wid  = tid >> 5;
    const int lm   = lane & 15;
    const int half = lane >> 4;
    const int wm   = wid >> 2;          // 0..1
    const int wn   = wid & 3;           // 0..3
    const int mb   = blockIdx.y * 128;  // block row base
    const int m0   = wm * 64;           // wave row base within tile
    const int n0   = blockIdx.x * 128 + wn * 32;

    // staging assignment: thread -> (row, 16-col segment), 2 x b128 each
    const int srow = tid >> 1;
    const int scol = (tid & 1) * 16;
    const u16* sg  = X + (size_t)(mb + srow) * K + scol;

    v8f acc[4][2];
#pragma unroll
    for (int i = 0; i < 4; i++)
#pragma unroll
        for (int j = 0; j < 2; j++) acc[i][j] = (v8f)0.0f;

    const u16* brow[2];
#pragma unroll
    for (int j = 0; j < 2; j++) brow[j] = W + (size_t)(n0 + 16 * j + lm) * K;

    // prologue: stage k0 = 0 into buffer 0
    {
        unsigned l = (unsigned)(uintptr_t)&Atile[0][srow][scol];
        async_copy16(l, sg);
        async_copy16(l + 16, sg + 8);
    }

    int buf = 0;
    for (int k0 = 0; k0 < K; k0 += 32, buf ^= 1) {
        __syncthreads();                // everyone done reading buf^1
        const bool more = (k0 + 32 < K);
        if (more) {                     // stage next tile into buf^1
            unsigned l = (unsigned)(uintptr_t)&Atile[buf ^ 1][srow][scol];
            const u16* g = sg + k0 + 32;
            async_copy16(l, g);
            async_copy16(l + 16, g + 8);
            __builtin_prefetch(brow[0] + k0 + 64, 0, 0);   // global_prefetch_b8
        }
        if (more) asm volatile("s_wait_asynccnt 2" ::: "memory");
        else      asm volatile("s_wait_asynccnt 0" ::: "memory");
        __syncthreads();                // current tile visible to all waves

        v16bf a[4], b[2];
#pragma unroll
        for (int i = 0; i < 4; i++) {
            const u16* ar = &Atile[buf][m0 + 16 * i + lm][0];
            a[i] = ld2(ar + 16 * half, ar + 16 * half + 8);
        }
#pragma unroll
        for (int j = 0; j < 2; j++)
            b[j] = ld2(brow[j] + k0 + 16 * half, brow[j] + k0 + 16 * half + 8);
#pragma unroll
        for (int i = 0; i < 4; i++)
#pragma unroll
            for (int j = 0; j < 2; j++)
                acc[i][j] = wmma_bf16(a[i], b[j], acc[i][j]);
    }

    // Epilogue: C layout -> lane holds col n0+16j+lm, rows mb+m0+16i + r + 8*half.
#pragma unroll
    for (int i = 0; i < 4; i++) {
#pragma unroll
        for (int j = 0; j < 2; j++) {
            const int col = n0 + 16 * j + lm;
            const float bv = bias ? bias[col] : 0.0f;
#pragma unroll
            for (int r = 0; r < 8; r++) {
                const int row = mb + m0 + 16 * i + r + 8 * half;
                float v = acc[i][j][r] + bv;
                if (RELU) v = fmaxf(v, 0.0f);
                const size_t idx = (size_t)row * Nout + col;
                if (res)  v += res[idx];
                if (outf) outf[idx] = v;
                if (outb) outb[idx] = f2bf(v);
            }
        }
    }
}

// ---------------------------------------------------------------------------
// Flash attention over bf16 Q/K/V stored as (B*N, A) with head h at cols
// [h*64, h*64+64). Block = 128 threads (4 waves); block does a (b, h, 64-query)
// tile; each wave owns 16 query rows; loop over 32-key chunks. Scores and
// probabilities never touch global memory (would be ~1GB of N x N fp32).
// ---------------------------------------------------------------------------
__global__ __launch_bounds__(128)
void flash_attn(const u16* __restrict__ Q, const u16* __restrict__ Kmat,
                const u16* __restrict__ V, const unsigned char* __restrict__ mask,
                u16* __restrict__ O) {
    __shared__ u16 Vt[64][40];          // V chunk transposed: Vt[d][key], padded row
    __shared__ u16 Pl[4][16][32];       // per-wave probability tile (q x key)

    const int tid  = threadIdx.x;
    const int lane = tid & 31;
    const int wid  = tid >> 5;          // 0..3
    const int lm   = lane & 15;
    const int half = lane >> 4;
    const int b    = blockIdx.z;
    const int h    = blockIdx.y;
    const int q0   = blockIdx.x * 64 + wid * 16;

    // Q fragments for d-chunks {0..31}, {32..63}; resident all kernel.
    const u16* qrow = Q + (size_t)(b * N_ + q0 + lm) * A_ + h * HD_;
    v16bf qa[2];
#pragma unroll
    for (int c = 0; c < 2; c++)
        qa[c] = ld2(qrow + 32 * c + 16 * half, qrow + 32 * c + 16 * half + 8);

    v8f acc[4];
#pragma unroll
    for (int t = 0; t < 4; t++) acc[t] = (v8f)0.0f;
    float mr[8], lr[8];
#pragma unroll
    for (int r = 0; r < 8; r++) { mr[r] = -__builtin_inff(); lr[r] = 0.0f; }

    for (int kv0 = 0; kv0 < N_; kv0 += 32) {
        __syncthreads();                // previous Vt fully consumed
        {   // cooperative V stage + transpose: 128 thr x 16 elements = 32x64
            const int key  = tid >> 2;
            const int dseg = (tid & 3) * 16;
            const u16* vr = V + (size_t)(b * N_ + kv0 + key) * A_ + h * HD_ + dseg;
            FragU f;
            f.u[0] = *(const uint4*)vr;
            f.u[1] = *(const uint4*)(vr + 8);
            const u16* e = (const u16*)&f;
#pragma unroll
            for (int xx = 0; xx < 16; xx++) Vt[dseg + xx][key] = e[xx];
        }
        __syncthreads();

        // Scores: s0 = keys kv0..+15, s1 = keys kv0+16..+31 (contract d=64 in 2 steps)
        v8f s0 = (v8f)0.0f, s1 = (v8f)0.0f;
#pragma unroll
        for (int c = 0; c < 2; c++) {
            const u16* k0p = Kmat + (size_t)(b * N_ + kv0 + lm) * A_ + h * HD_ + 32 * c + 16 * half;
            const u16* k1p = Kmat + (size_t)(b * N_ + kv0 + 16 + lm) * A_ + h * HD_ + 32 * c + 16 * half;
            v16bf kb0 = ld2(k0p, k0p + 8);
            v16bf kb1 = ld2(k1p, k1p + 8);
            s0 = wmma_bf16(qa[c], kb0, s0);
            s1 = wmma_bf16(qa[c], kb1, s1);
        }
        s0 *= SCALE_;
        s1 *= SCALE_;

        // Mask (B,1,N,N) bool
#pragma unroll
        for (int r = 0; r < 8; r++) {
            const unsigned char* mp =
                mask + ((size_t)b * N_ + (q0 + r + 8 * half)) * N_ + kv0;
            if (mp[lm])      s0[r] = -__builtin_inff();
            if (mp[16 + lm]) s1[r] = -__builtin_inff();
        }

        // Online softmax. Row r+8*half lives in lanes of one 16-lane half;
        // xor masks 1,2,4,8 reduce within the half.
        float p0[8], p1[8];
#pragma unroll
        for (int r = 0; r < 8; r++) {
            float cm = fmaxf(s0[r], s1[r]);
#pragma unroll
            for (int sh = 1; sh < 16; sh <<= 1) cm = fmaxf(cm, __shfl_xor(cm, sh, 32));
            const float nm = fmaxf(mr[r], cm);
            const float sc = (mr[r] == nm) ? 1.0f : __expf(mr[r] - nm);
            mr[r] = nm;
            p0[r] = __expf(s0[r] - nm);
            p1[r] = __expf(s1[r] - nm);
            float rsum = p0[r] + p1[r];
#pragma unroll
            for (int sh = 1; sh < 16; sh <<= 1) rsum += __shfl_xor(rsum, sh, 32);
            lr[r] = lr[r] * sc + rsum;
#pragma unroll
            for (int t = 0; t < 4; t++) acc[t][r] *= sc;
        }

        // Round P to bf16 into per-wave LDS tile (row-major q x key_local)
#pragma unroll
        for (int r = 0; r < 8; r++) {
            Pl[wid][r + 8 * half][lm]      = f2bf(p0[r]);
            Pl[wid][r + 8 * half][16 + lm] = f2bf(p1[r]);
        }
        // same-wave LDS write -> read fence
        asm volatile("s_wait_dscnt 0" ::: "memory");

        // P as A-fragment, V^T rows as B-fragments; O += P @ V
        v16bf pA = ld2(&Pl[wid][lm][16 * half], &Pl[wid][lm][16 * half + 8]);
#pragma unroll
        for (int t = 0; t < 4; t++) {
            const u16* vp = &Vt[16 * t + lm][16 * half];
            v16bf vb = ld2(vp, vp + 8);
            acc[t] = wmma_bf16(pA, vb, acc[t]);
        }
    }

    // Normalize and write O (bf16) back into (B*N, A) layout.
#pragma unroll
    for (int r = 0; r < 8; r++) {
        const float inv = 1.0f / lr[r];
        u16* orow = O + (size_t)(b * N_ + q0 + r + 8 * half) * A_ + h * HD_;
#pragma unroll
        for (int t = 0; t < 4; t++)
            orow[16 * t + lm] = f2bf(acc[t][r] * inv);
    }
}

// ---------------------------------------------------------------------------
// Launch pipeline
// ---------------------------------------------------------------------------
extern "C" void kernel_launch(void* const* d_in, const int* in_sizes, int n_in,
                              void* d_out, int out_size, void* d_ws, size_t ws_size,
                              hipStream_t stream) {
    (void)in_sizes; (void)n_in; (void)out_size; (void)ws_size;

    const float*         x    = (const float*)d_in[0];
    const unsigned char* mask = (const unsigned char*)d_in[1];
    const float* wq = (const float*)d_in[2];  const float* bq = (const float*)d_in[3];
    const float* wk = (const float*)d_in[4];  const float* bk = (const float*)d_in[5];
    const float* wv = (const float*)d_in[6];  const float* bv = (const float*)d_in[7];
    const float* wo = (const float*)d_in[8];  const float* bo = (const float*)d_in[9];
    const float* w1 = (const float*)d_in[10]; const float* b1 = (const float*)d_in[11];
    const float* w2 = (const float*)d_in[12]; const float* b2 = (const float*)d_in[13];
    const float* ln1g = (const float*)d_in[14]; const float* ln1b = (const float*)d_in[15];
    const float* ln2g = (const float*)d_in[16]; const float* ln2b = (const float*)d_in[17];
    float* out = (float*)d_out;

    char* wp = (char*)d_ws;
    auto alloc = [&](size_t bytes) -> char* {
        char* p = wp;
        wp += (bytes + 255) & ~(size_t)255;
        return p;
    };
    u16*   hln = (u16*)alloc((size_t)M_ * D_ * 2);
    u16*   wqb = (u16*)alloc((size_t)A_ * D_ * 2);
    u16*   wkb = (u16*)alloc((size_t)A_ * D_ * 2);
    u16*   wvb = (u16*)alloc((size_t)A_ * D_ * 2);
    u16*   wob = (u16*)alloc((size_t)D_ * A_ * 2);
    u16*   w1b = (u16*)alloc((size_t)DFF_ * D_ * 2);
    u16*   w2b = (u16*)alloc((size_t)D_ * DFF_ * 2);
    u16*   Qb  = (u16*)alloc((size_t)M_ * A_ * 2);
    u16*   Kb  = (u16*)alloc((size_t)M_ * A_ * 2);
    u16*   Vb  = (u16*)alloc((size_t)M_ * A_ * 2);
    u16*   Ob  = (u16*)alloc((size_t)M_ * A_ * 2);
    float* x2  = (float*)alloc((size_t)M_ * D_ * 4);
    u16*   h2  = (u16*)alloc((size_t)M_ * D_ * 2);
    u16*   m1  = (u16*)alloc((size_t)M_ * DFF_ * 2);

    // 1) weights fp32 -> bf16
    auto cvt = [&](const float* src, u16* dst, int n) {
        cvt_bf16<<<n / 1024, 256, 0, stream>>>(src, dst, n);
    };
    cvt(wq, wqb, A_ * D_);
    cvt(wk, wkb, A_ * D_);
    cvt(wv, wvb, A_ * D_);
    cvt(wo, wob, D_ * A_);
    cvt(w1, w1b, DFF_ * D_);
    cvt(w2, w2b, D_ * DFF_);

    // 2) ln1
    ln_bf16<<<M_, 256, 0, stream>>>(x, ln1g, ln1b, hln);

    // 3) Q/K/V projections (bf16 outputs)
    dim3 blk(256);
    dim3 gQKV(A_ / 128, M_ / 128);
    gemm_wmma<false><<<gQKV, blk, 0, stream>>>(hln, wqb, bq, nullptr, nullptr, Qb, M_, A_, D_);
    gemm_wmma<false><<<gQKV, blk, 0, stream>>>(hln, wkb, bk, nullptr, nullptr, Kb, M_, A_, D_);
    gemm_wmma<false><<<gQKV, blk, 0, stream>>>(hln, wvb, bv, nullptr, nullptr, Vb, M_, A_, D_);

    // 4) flash attention
    flash_attn<<<dim3(N_ / 64, H_, B_), 128, 0, stream>>>(Qb, Kb, Vb, mask, Ob);

    // 5) output projection + residual -> x2 (fp32)
    gemm_wmma<false><<<dim3(D_ / 128, M_ / 128), blk, 0, stream>>>(
        Ob, wob, bo, x, x2, nullptr, M_, D_, A_);

    // 6) ln2
    ln_bf16<<<M_, 256, 0, stream>>>(x2, ln2g, ln2b, h2);

    // 7) MLP up + ReLU (bf16 out)
    gemm_wmma<true><<<dim3(DFF_ / 128, M_ / 128), blk, 0, stream>>>(
        h2, w1b, b1, nullptr, nullptr, m1, M_, DFF_, D_);

    // 8) MLP down + residual -> final fp32 output
    gemm_wmma<false><<<dim3(D_ / 128, M_ / 128), blk, 0, stream>>>(
        m1, w2b, b2, x2, out, nullptr, M_, D_, DFF_);
}
